// Attention_29291676958747
// MI455X (gfx1250) — matrix operand order repaired and checked by reading, not compiled, not run on hardware
//
#include <hip/hip_runtime.h>
#include <hip/hip_bf16.h>

// ---------------------------------------------------------------------------
// CDNA5 (gfx1250) attention pipeline, wave32 + v_wmma_f32_16x16x32_f16.
// Register-blocked: GEMMs 2Mx4N per wave, attention 2 i-tiles per wave.
//
// Shapes: B=2, DIM=512, HEADS=8, DH=64, N=2048, DPB_DIM=256, TABLE=2N-1=4095
// SCALE = DH^-0.5 = 0.125
// ---------------------------------------------------------------------------

typedef __attribute__((ext_vector_type(8)))  _Float16 v8h;
typedef __attribute__((ext_vector_type(16))) _Float16 v16h;
typedef __attribute__((ext_vector_type(8)))  float    v8f;

#define NB      2
#define NHEADS  8
#define DH      64
#define NSEQ    2048
#define DIMC    512
#define O3      1536
#define DPBD    256
#define TBL     4095

// ---- WMMA fragment helpers (wave32 layouts per ISA 7.12.2) -----------------
// A (16x32 f16, MxK): lane m=lane&15, hi=lane>>4; holds K = hi*8+{0..7} in
// VGPR0-3 and K = 16+hi*8+{0..7} in VGPR4-7.  Source row-major [M][K].
__device__ __forceinline__ v16h load_a_frag(const _Float16* __restrict__ base,
                                            int ld, int lane) {
  int m = lane & 15, hi = lane >> 4;
  const _Float16* p = base + (size_t)m * ld + hi * 8;
  v8h lo = *(const v8h*)p;
  v8h hh = *(const v8h*)(p + 16);
  return __builtin_shufflevector(lo, hh, 0,1,2,3,4,5,6,7,8,9,10,11,12,13,14,15);
}
// B (32x16 f16, KxN): lane n=lane&15, hi=lane>>4; holds K = hi*8+{0..7} and
// K = 16+hi*8+{0..7} at fixed column n.  Source is B^T row-major, i.e. memory [N][K], K contiguous.
__device__ __forceinline__ v16h load_b_frag(const _Float16* __restrict__ base,
                                            int ld, int lane) {
  int n = lane & 15, hi = lane >> 4;
  const _Float16* p = base + (size_t)n * ld + hi * 8;
  v8h b0 = *(const v8h*)p;
  v8h b1 = *(const v8h*)(p + 16);
  return __builtin_shufflevector(b0, b1, 0,1,2,3,4,5,6,7,8,9,10,11,12,13,14,15);
}
__device__ __forceinline__ v8f wmma_f16(v16h a, v16h b, v8f c) {
  // D = A x B + C, f32 accumulate.  (neg_a, A, neg_b, B, c_mod, C, reuseA, reuseB)
  return __builtin_amdgcn_wmma_f32_16x16x32_f16(false, a, false, b, (short)0, c,
                                                false, false);
}
// C layout: element e of the v8f accumulator is (M = e + 8*(lane>>4), N = lane&15).

// 16-lane (half-wave) row reductions across the N dimension of a C fragment.
__device__ __forceinline__ float rowmax16(float x) {
  #pragma unroll
  for (int m = 1; m < 16; m <<= 1) x = fmaxf(x, __shfl_xor(x, m, 32));
  return x;
}
__device__ __forceinline__ float rowsum16(float x) {
  #pragma unroll
  for (int m = 1; m < 16; m <<= 1) x += __shfl_xor(x, m, 32);
  return x;
}

// ---------------------------------------------------------------------------
// Prep kernels
// ---------------------------------------------------------------------------
__global__ void cvt_f32_f16(const float* __restrict__ s, _Float16* __restrict__ d,
                            int n) {
  int i = blockIdx.x * blockDim.x + threadIdx.x;
  if (i < n) d[i] = (_Float16)s[i];
}

// x [b][c][n] f32  ->  xT [b][n][c] f16  (B operand for QKV GEMM)
__global__ void transpose_x(const float* __restrict__ x, _Float16* __restrict__ xT) {
  int i = blockIdx.x * blockDim.x + threadIdx.x;
  if (i >= NB * DIMC * NSEQ) return;
  int n = i & (NSEQ - 1);
  int c = (i >> 11) & (DIMC - 1);
  int b = i >> 20;
  xT[((size_t)b * NSEQ + n) * DIMC + c] = (_Float16)x[i];
}

// Dynamic position bias, stored transposed: tableT[h][idx], idx in [0, 4095)
// so that attention's per-lane bias reads are 8 consecutive floats.
__global__ void dpb_kernel(const float* __restrict__ w1, const float* __restrict__ b1,
                           const float* __restrict__ w2, const float* __restrict__ b2,
                           float* __restrict__ tableT) {
  int idx = blockIdx.x * blockDim.x + threadIdx.x;
  if (idx >= TBL) return;
  float pos = (float)(idx - (NSEQ - 1));
  float acc[NHEADS];
  #pragma unroll
  for (int h = 0; h < NHEADS; h++) acc[h] = b2[h];
  for (int c = 0; c < DPBD; c++) {
    float hv = pos * w1[c] + b1[c];
    hv = hv / (1.0f + __expf(-hv));          // silu
    #pragma unroll
    for (int h = 0; h < NHEADS; h++) acc[h] += hv * w2[h * DPBD + c];
  }
  #pragma unroll
  for (int h = 0; h < NHEADS; h++) tableT[(size_t)h * TBL + idx] = acc[h];
}

// ---------------------------------------------------------------------------
// QKV GEMM: qkv[o,n] = sum_c w[o,c] * x[c,n]   (per batch; o in [0,1536))
//   2Mx4N register blocking: 8 WMMAs per K-chunk from 4 A + 8 B loads.
//   q,k -> [b][h][n][d] f16 (q pre-scaled by 0.125), v -> [b][h][d][n] f16
// ---------------------------------------------------------------------------
__global__ __launch_bounds__(128, 1)
void qkv_gemm(const _Float16* __restrict__ w, const _Float16* __restrict__ xT,
              _Float16* __restrict__ q, _Float16* __restrict__ k,
              _Float16* __restrict__ v) {
  int lane = threadIdx.x & 31, wave = threadIdx.x >> 5;
  int b = blockIdx.z;
  int ot = blockIdx.x * 32;                     // 32 output rows per WG
  int nb = blockIdx.y * 256 + wave * 64;        // 64 columns per wave
  const _Float16* xb = xT + (size_t)b * NSEQ * DIMC;
  v8f acc[2][4] = {};
  for (int kc = 0; kc < DIMC; kc += 32) {
    __builtin_prefetch(w + (size_t)ot * DIMC + kc + 64, 0, 0);
    v16h a0 = load_a_frag(w + (size_t)ot * DIMC + kc, DIMC, lane);
    v16h a1 = load_a_frag(w + (size_t)(ot + 16) * DIMC + kc, DIMC, lane);
    #pragma unroll
    for (int t = 0; t < 4; t++) {
      v16h bb = load_b_frag(xb + (size_t)(nb + t * 16) * DIMC + kc, DIMC, lane);
      acc[0][t] = wmma_f16(a0, bb, acc[0][t]);
      acc[1][t] = wmma_f16(a1, bb, acc[1][t]);
    }
  }
  int hi = lane >> 4, nl = lane & 15;
  #pragma unroll
  for (int mt = 0; mt < 2; mt++) {
    int ot2 = ot + mt * 16;
    int region = ot2 >> 9;               // 0=q, 1=k, 2=v
    int h = (ot2 >> 6) & 7;
    int d0 = (ot2 & 63) + 8 * hi;        // multiple of 8 -> 16B aligned stores
    size_t bh = (size_t)(b * NHEADS + h);
    #pragma unroll
    for (int t = 0; t < 4; t++) {
      int n = nb + t * 16 + nl;
      if (region == 2) {
        #pragma unroll
        for (int e = 0; e < 8; e++)
          v[(bh * DH + d0 + e) * NSEQ + n] = (_Float16)acc[mt][t][e];
      } else {
        float sc = (region == 0) ? 0.125f : 1.0f;   // SCALE on q
        _Float16* dstp = ((region == 0) ? q : k) + (bh * NSEQ + n) * DH + d0;
        v8h pk;
        #pragma unroll
        for (int e = 0; e < 8; e++) pk[e] = (_Float16)(acc[mt][t][e] * sc);
        *(v8h*)dstp = pk;
      }
    }
  }
}

// ---------------------------------------------------------------------------
// Attention with online softmax.  One wave = two 16-row query tiles (32 rows),
// so K and V fragments are loaded once per j-chunk and reused: 16 WMMAs per
// 16 global b128 loads.
//   q,k: [b][h][n][d], v: [b][h][d][n]; out -> attnT [b][n][c=h*64+d] f16
// ---------------------------------------------------------------------------
__global__ __launch_bounds__(128, 1)
void attn_kernel(const _Float16* __restrict__ qg,
                 const _Float16* __restrict__ kg,
                 const _Float16* __restrict__ vg,
                 const float* __restrict__ tableT,
                 _Float16* __restrict__ attnT) {
  __shared__ __align__(16) _Float16 pbuf[4][32 * 32];  // per-wave P bounce (2KB)
  int lane = threadIdx.x & 31, wave = threadIdx.x >> 5;
  int h = blockIdx.y, b = blockIdx.z;
  int it0 = blockIdx.x * 128 + wave * 32;
  size_t bh = (size_t)(b * NHEADS + h);
  const _Float16* qh = qg + bh * (size_t)(NSEQ * DH);
  const _Float16* kh = kg + bh * (size_t)(NSEQ * DH);
  const _Float16* vh = vg + bh * (size_t)(DH * NSEQ);
  const float* tb = tableT + (size_t)h * TBL;
  int hi = lane >> 4, nl = lane & 15;

  v16h aQ[2][2];
  #pragma unroll
  for (int m = 0; m < 2; m++) {
    aQ[m][0] = load_a_frag(qh + (size_t)(it0 + m * 16) * DH, DH, lane);
    aQ[m][1] = load_a_frag(qh + (size_t)(it0 + m * 16) * DH + 32, DH, lane);
  }

  v8f o[2][4] = {};
  float rm[2][8], rs[2][8];
  #pragma unroll
  for (int m = 0; m < 2; m++)
    #pragma unroll
    for (int e = 0; e < 8; e++) { rm[m][e] = -1e30f; rs[m][e] = 0.0f; }
  _Float16* pb = &pbuf[wave][0];

  for (int jc = 0; jc < NSEQ; jc += 32) {
    // K fragments for two 16-col j-tiles, shared by both i-tiles
    v16h bK00 = load_b_frag(kh + (size_t)jc * DH, DH, lane);
    v16h bK01 = load_b_frag(kh + (size_t)jc * DH + 32, DH, lane);
    v16h bK10 = load_b_frag(kh + (size_t)(jc + 16) * DH, DH, lane);
    v16h bK11 = load_b_frag(kh + (size_t)(jc + 16) * DH + 32, DH, lane);

    #pragma unroll
    for (int m = 0; m < 2; m++) {
      v8f s0 = {}, s1 = {};
      s0 = wmma_f16(aQ[m][0], bK00, s0);
      s0 = wmma_f16(aQ[m][1], bK01, s0);
      s1 = wmma_f16(aQ[m][0], bK10, s1);
      s1 = wmma_f16(aQ[m][1], bK11, s1);

      // + dynamic position bias from tableT[h][i - j + N - 1]; per lane the
      // 8 reads (e -> i+e) are consecutive floats -> merged wide loads.
      const float* tp = tb + (it0 + m * 16 + 8 * hi - (jc + nl) + NSEQ - 1);
      #pragma unroll
      for (int e = 0; e < 8; e++) {
        s0[e] += tp[e];
        s1[e] += tp[e - 16];
      }

      // online softmax update + write P (f16) to LDS row-major [32][32]
      #pragma unroll
      for (int e = 0; e < 8; e++) {
        float tmax = fmaxf(rowmax16(s0[e]), rowmax16(s1[e]));
        float nm = fmaxf(rm[m][e], tmax);
        float scale = __expf(rm[m][e] - nm);
        float p0 = __expf(s0[e] - nm);
        float p1 = __expf(s1[e] - nm);
        rs[m][e] = rs[m][e] * scale + rowsum16(p0) + rowsum16(p1);
        rm[m][e] = nm;
        o[m][0][e] *= scale; o[m][1][e] *= scale;
        o[m][2][e] *= scale; o[m][3][e] *= scale;
        int row = m * 16 + e + 8 * hi;
        pb[row * 32 + nl]      = (_Float16)p0;
        pb[row * 32 + 16 + nl] = (_Float16)p1;
      }
    }
    // C-layout -> A-layout bounce: drain LDS stores before re-reading.
    asm volatile("s_wait_dscnt 0x0" ::: "memory");
    v16h aP0 = load_a_frag(pb, 32, lane);
    v16h aP1 = load_a_frag(pb + 16 * 32, 32, lane);

    // O += P . V^T  (V stored [d][n]); each V fragment reused by both i-tiles
    #pragma unroll
    for (int t = 0; t < 4; t++) {
      v16h bV = load_b_frag(vh + (size_t)(t * 16) * NSEQ + jc, NSEQ, lane);
      o[0][t] = wmma_f16(aP0, bV, o[0][t]);
      o[1][t] = wmma_f16(aP1, bV, o[1][t]);
    }
    // (LDS per-wave in-order guarantees WAR safety for next iteration's stores)
  }

  // normalize rows and store transposed: attnT[b][n][h*64 + d]
  _Float16* dst = attnT + (size_t)b * NSEQ * DIMC;
  #pragma unroll
  for (int m = 0; m < 2; m++)
    #pragma unroll
    for (int e = 0; e < 8; e++) {
      float inv = 1.0f / rs[m][e];
      int n = it0 + m * 16 + e + 8 * hi;
      int cb = h * DH + nl;
      dst[(size_t)n * DIMC + cb + 0]  = (_Float16)(o[m][0][e] * inv);
      dst[(size_t)n * DIMC + cb + 16] = (_Float16)(o[m][1][e] * inv);
      dst[(size_t)n * DIMC + cb + 32] = (_Float16)(o[m][2][e] * inv);
      dst[(size_t)n * DIMC + cb + 48] = (_Float16)(o[m][3][e] * inv);
    }
}

// ---------------------------------------------------------------------------
// Out projection: out[b,o,n] = sum_c w_out[o,c] * attn[c,n] + b_out[o]
// 2Mx4N register blocking, f32 output.
// ---------------------------------------------------------------------------
__global__ __launch_bounds__(128, 1)
void proj_gemm(const _Float16* __restrict__ w, const _Float16* __restrict__ aT,
               const float* __restrict__ bias, float* __restrict__ out) {
  int lane = threadIdx.x & 31, wave = threadIdx.x >> 5;
  int b = blockIdx.z;
  int ot = blockIdx.x * 32;
  int nb = blockIdx.y * 256 + wave * 64;
  const _Float16* ab = aT + (size_t)b * NSEQ * DIMC;
  v8f acc[2][4] = {};
  for (int kc = 0; kc < DIMC; kc += 32) {
    __builtin_prefetch(w + (size_t)ot * DIMC + kc + 64, 0, 0);
    v16h a0 = load_a_frag(w + (size_t)ot * DIMC + kc, DIMC, lane);
    v16h a1 = load_a_frag(w + (size_t)(ot + 16) * DIMC + kc, DIMC, lane);
    #pragma unroll
    for (int t = 0; t < 4; t++) {
      v16h bb = load_b_frag(ab + (size_t)(nb + t * 16) * DIMC + kc, DIMC, lane);
      acc[0][t] = wmma_f16(a0, bb, acc[0][t]);
      acc[1][t] = wmma_f16(a1, bb, acc[1][t]);
    }
  }
  int hi = lane >> 4, nl = lane & 15;
  #pragma unroll
  for (int mt = 0; mt < 2; mt++)
    #pragma unroll
    for (int t = 0; t < 4; t++) {
      int n = nb + t * 16 + nl;
      #pragma unroll
      for (int e = 0; e < 8; e++) {
        int o = ot + mt * 16 + e + 8 * hi;
        out[(size_t)b * DIMC * NSEQ + (size_t)o * NSEQ + n] = acc[mt][t][e] + bias[o];
      }
    }
}

// ---------------------------------------------------------------------------
// Launch
// ---------------------------------------------------------------------------
extern "C" void kernel_launch(void* const* d_in, const int* in_sizes, int n_in,
                              void* d_out, int out_size, void* d_ws, size_t ws_size,
                              hipStream_t stream) {
  const float* x      = (const float*)d_in[0];
  const float* w_qkv  = (const float*)d_in[1];
  const float* w_out  = (const float*)d_in[2];
  const float* b_out  = (const float*)d_in[3];
  const float* w1_dpb = (const float*)d_in[4];
  const float* b1_dpb = (const float*)d_in[5];
  const float* w2_dpb = (const float*)d_in[6];
  const float* b2_dpb = (const float*)d_in[7];
  // d_in[8] = use_dist (0 in this configuration)
  float* out = (float*)d_out;

  char* ws = (char*)d_ws;
  constexpr size_t OFF_XT   = 0;                                     // 4 MB
  constexpr size_t OFF_WQKV = OFF_XT   + (size_t)NB * NSEQ * DIMC * 2;
  constexpr size_t OFF_WOUT = OFF_WQKV + (size_t)O3 * DIMC * 2;
  constexpr size_t OFF_Q    = OFF_WOUT + (size_t)DIMC * DIMC * 2;
  constexpr size_t OFF_K    = OFF_Q    + (size_t)NB * NHEADS * NSEQ * DH * 2;
  constexpr size_t OFF_V    = OFF_K    + (size_t)NB * NHEADS * NSEQ * DH * 2;
  constexpr size_t OFF_AT   = OFF_V    + (size_t)NB * NHEADS * NSEQ * DH * 2;
  constexpr size_t OFF_TBL  = OFF_AT   + (size_t)NB * NSEQ * DIMC * 2;

  _Float16* xT    = (_Float16*)(ws + OFF_XT);
  _Float16* wqh   = (_Float16*)(ws + OFF_WQKV);
  _Float16* woh   = (_Float16*)(ws + OFF_WOUT);
  _Float16* qh    = (_Float16*)(ws + OFF_Q);
  _Float16* kh    = (_Float16*)(ws + OFF_K);
  _Float16* vh    = (_Float16*)(ws + OFF_V);
  _Float16* attnT = (_Float16*)(ws + OFF_AT);
  float*    table = (float*)   (ws + OFF_TBL);

  // prep
  {
    int n1 = O3 * DIMC;
    cvt_f32_f16<<<(n1 + 255) / 256, 256, 0, stream>>>(w_qkv, wqh, n1);
    int n2 = DIMC * DIMC;
    cvt_f32_f16<<<(n2 + 255) / 256, 256, 0, stream>>>(w_out, woh, n2);
    int n3 = NB * DIMC * NSEQ;
    transpose_x<<<(n3 + 255) / 256, 256, 0, stream>>>(x, xT);
    dpb_kernel<<<(TBL + 127) / 128, 128, 0, stream>>>(w1_dpb, b1_dpb, w2_dpb,
                                                      b2_dpb, table);
  }
  // qkv: grid (1536/32, 2048/256, B), 4 waves per WG, 2Mx4N per wave
  qkv_gemm<<<dim3(O3 / 32, NSEQ / 256, NB), 128, 0, stream>>>(wqh, xT, qh, kh, vh);
  // attention: grid (2048/128, H, B), 4 waves per WG (two 16-row tiles each)
  attn_kernel<<<dim3(NSEQ / 128, NHEADS, NB), 128, 0, stream>>>(qh, kh, vh, table,
                                                                attnT);
  // projection: grid (512/32, 2048/256, B)
  proj_gemm<<<dim3(DIMC / 32, NSEQ / 256, NB), 128, 0, stream>>>(woh, attnT, b_out,
                                                                 out);
}
